// GraphSageDQN_88424786690660
// MI455X (gfx1250) — matrix operand order repaired and checked by reading, not compiled, not run on hardware
//
#include <hip/hip_runtime.h>
#include <hip/hip_bf16.h>

// Problem constants (match reference)
#define N_NODES 100000
#define N_EDGES 1600000
#define DIM     128
#define ACT     512
#define NGRAPH  64

typedef __attribute__((ext_vector_type(16))) __bf16 v16bf;
typedef __attribute__((ext_vector_type(8)))  float  v8f;
typedef __attribute__((ext_vector_type(4)))  int    v4i;

union V16BF { v16bf v; uint4 q[2]; };

#define CEILDIV(a, b) (((a) + (b) - 1) / (b))

// LDS row stride for staged W2 (bf16 elems): 136*2B = 272B = 68 dwords per row,
// so lane r (row stride) maps to bank 4r mod 64 -> conflict-free b128 reads.
#define LDS_STRIDE 136

#if defined(__gfx1250__) && __has_builtin(__builtin_amdgcn_global_load_async_to_lds_b128)
#define HAVE_ASYNC_LDS 1
#else
#define HAVE_ASYNC_LDS 0
#endif

// ---------------- degree / normalization ----------------

__global__ void k_init_deg(float* deg) {
    int i = blockIdx.x * blockDim.x + threadIdx.x;
    if (i < N_NODES) deg[i] = 1.0f;              // self-loop contributes 1
}

__global__ void k_deg(const int* dst, float* deg) {
    int e = blockIdx.x * blockDim.x + threadIdx.x;
    if (e < N_EDGES) atomicAdd(&deg[dst[e]], 1.0f);
}

__global__ void k_dis(float* deg) {
    int i = blockIdx.x * blockDim.x + threadIdx.x;
    if (i < N_NODES) deg[i] = rsqrtf(deg[i]);    // deg >= 1 always
}

// ---------------- GEMM1: h = x @ W1  (K = 3, scalar) ----------------

__global__ void k_lin1(const float* __restrict__ x, const float* __restrict__ W1,
                       float* __restrict__ h) {
    int t = blockIdx.x * blockDim.x + threadIdx.x;
    if (t >= N_NODES * DIM) return;
    int i = t >> 7, f = t & (DIM - 1);
    float x0 = x[i * 3 + 0], x1 = x[i * 3 + 1], x2 = x[i * 3 + 2];
    h[t] = x0 * W1[f] + x1 * W1[DIM + f] + x2 * W1[2 * DIM + f];
}

// ---------------- aggregation: self-loop init + edge scatter ----------------

__global__ void k_selfinit(const float* __restrict__ h, const float* __restrict__ dis,
                           float* __restrict__ agg) {
    int t = blockIdx.x * blockDim.x + threadIdx.x;
    if (t >= N_NODES * DIM) return;
    int i = t >> 7;
    float d = dis[i];
    agg[t] = h[t] * d * d;
}

// One wave per edge; e is wave-uniform -> edge endpoints load via scalar path.
__global__ void __launch_bounds__(256)
k_scatter(const int* __restrict__ src, const int* __restrict__ dst,
          const float* __restrict__ dis, const float* __restrict__ h,
          float* __restrict__ agg) {
    int e = blockIdx.x * 8 + (threadIdx.x >> 5);   // 8 waves per block
    int lane = threadIdx.x & 31;
    if (e >= N_EDGES) return;
    int s = src[e], d = dst[e];
    float w = dis[s] * dis[d];
    const float4* hp = (const float4*)(h + (long)s * DIM) + lane;   // coalesced 512B/wave
    float4 v = *hp;
    float* ap = agg + (long)d * DIM + lane * 4;
    atomicAdd(ap + 0, v.x * w);
    atomicAdd(ap + 1, v.y * w);
    atomicAdd(ap + 2, v.z * w);
    atomicAdd(ap + 3, v.w * w);
}

// ---------------- bias + relu + bf16 convert ----------------

__global__ void k_relu_bf16(const float* __restrict__ agg, const float* __restrict__ b,
                            __bf16* __restrict__ out) {
    int t = blockIdx.x * blockDim.x + threadIdx.x;
    if (t >= N_NODES * DIM) return;
    int f = t & (DIM - 1);
    float v = agg[t] + b[f];
    out[t] = (__bf16)(v > 0.0f ? v : 0.0f);
}

// ---------------- pack W2 -> bf16, transposed [n][k] ----------------

__global__ void k_packW2(const float* __restrict__ W2, __bf16* __restrict__ W2b) {
    int t = blockIdx.x * blockDim.x + threadIdx.x;
    if (t >= DIM * DIM) return;
    int n = t >> 7, k = t & (DIM - 1);
    W2b[n * DIM + k] = (__bf16)W2[k * DIM + n];
}

// ---------------- GEMM2: C[N,128] = A[N,128] @ W2 via WMMA bf16 ----------------
// W2 (bf16, transposed) staged once per block into padded LDS via async-to-LDS,
// B fragments fed by ds_load_b128; one wave owns a 16-row strip x 128 cols.

__global__ void __launch_bounds__(256)
k_gemm_wmma(const __bf16* __restrict__ A, const __bf16* __restrict__ Bt,
            float* __restrict__ C) {
    __shared__ __align__(16) __bf16 lw[DIM * LDS_STRIDE];

    // Cooperative copy: 128 rows x 16 chunks of 16B -> 8 chunks per thread.
    {
        int tid = threadIdx.x;
        for (int c = tid; c < DIM * 16; c += 256) {
            int row = c >> 4, seg = c & 15;
            const __bf16* g = Bt + row * DIM + seg * 8;
            __bf16* l = &lw[row * LDS_STRIDE + seg * 8];
#if HAVE_ASYNC_LDS
            __builtin_amdgcn_global_load_async_to_lds_b128(
                (__attribute__((address_space(1))) v4i*)(size_t)g,
                (__attribute__((address_space(3))) v4i*)(unsigned)(size_t)l,
                0, 0);
#else
            *(uint4*)l = *(const uint4*)g;
#endif
        }
#if HAVE_ASYNC_LDS
#if __has_builtin(__builtin_amdgcn_s_wait_asynccnt)
        __builtin_amdgcn_s_wait_asynccnt(0);
#else
        asm volatile("s_wait_asynccnt 0" ::: "memory");
#endif
#endif
        __syncthreads();
    }

    int wave = blockIdx.x * 8 + (threadIdx.x >> 5);   // uniform per wave
    int lane = threadIdx.x & 31;
    if (wave >= N_NODES / 16) return;
    int row0 = wave * 16;
    int r = lane & 15;
    int half = lane >> 4;

    v8f zero = {0.f, 0.f, 0.f, 0.f, 0.f, 0.f, 0.f, 0.f};
    v8f acc[8];
#pragma unroll
    for (int j = 0; j < 8; ++j) acc[j] = zero;

#pragma unroll
    for (int ks = 0; ks < DIM; ks += 32) {
        // A fragment: 16-bit A 16x32 layout — elems 0..7 = K[kb..kb+7],
        // elems 8..15 = K[kb+16..kb+23], kb = ks + half*8
        V16BF a;
        const __bf16* ap = A + (long)(row0 + r) * DIM + ks + half * 8;
        a.q[0] = *(const uint4*)(ap);
        a.q[1] = *(const uint4*)(ap + 16);

        // Preload all 8 B fragments from LDS, then run the WMMA chain.
        V16BF bfrag[8];
#pragma unroll
        for (int j = 0; j < 8; ++j) {
            const __bf16* bp = &lw[(j * 16 + r) * LDS_STRIDE + ks + half * 8];
            bfrag[j].q[0] = *(const uint4*)(bp);
            bfrag[j].q[1] = *(const uint4*)(bp + 16);
        }
#pragma unroll
        for (int j = 0; j < 8; ++j) {
            acc[j] = __builtin_amdgcn_wmma_f32_16x16x32_bf16(
                false, a.v, false, bfrag[j].v, (short)0, acc[j], false, false);
        }
    }

    // D layout: VGPR m -> row (m + half*8), col = tile*16 + r
#pragma unroll
    for (int j = 0; j < 8; ++j) {
        float* cp = C + (long)(row0 + half * 8) * DIM + j * 16 + r;
#pragma unroll
        for (int m = 0; m < 8; ++m) cp[(long)m * DIM] = acc[j][m];
    }
}

// ---------------- pooling ----------------

__global__ void k_zero_pool(float* sums, float* cnts) {
    int t = blockIdx.x * blockDim.x + threadIdx.x;
    if (t < NGRAPH * DIM) sums[t] = 0.0f;
    if (t < NGRAPH) cnts[t] = 0.0f;
}

__global__ void k_pool(const float* __restrict__ agg, const float* __restrict__ b2,
                       const int* __restrict__ batch, float* sums, float* cnts) {
    int t = blockIdx.x * blockDim.x + threadIdx.x;
    if (t >= N_NODES * DIM) return;
    int i = t >> 7, f = t & (DIM - 1);
    int g = batch[i];
    atomicAdd(&sums[g * DIM + f], agg[t] + b2[f]);
    if (f == 0) atomicAdd(&cnts[g], 1.0f);
}

// ---------------- final GEMM: q[G,A] = (sums/cnt) @ Wf + bf ----------------

__global__ void k_final(const float* __restrict__ sums, const float* __restrict__ cnts,
                        const float* __restrict__ Wf, const float* __restrict__ bfv,
                        float* __restrict__ q) {
    int t = blockIdx.x * blockDim.x + threadIdx.x;
    if (t >= NGRAPH * ACT) return;
    int g = t / ACT, a = t % ACT;
    float acc = 0.0f;
    for (int k = 0; k < DIM; ++k) acc += sums[g * DIM + k] * Wf[k * ACT + a];
    float c = cnts[g];
    c = c > 1.0f ? c : 1.0f;
    q[t] = acc / c + bfv[a];
}

// ---------------- launch ----------------

extern "C" void kernel_launch(void* const* d_in, const int* in_sizes, int n_in,
                              void* d_out, int out_size, void* d_ws, size_t ws_size,
                              hipStream_t stream) {
    const float* x     = (const float*)d_in[0];
    const int*   eidx  = (const int*)d_in[1];     // [2, E]
    const int*   batch = (const int*)d_in[2];
    const float* W1    = (const float*)d_in[3];
    const float* b1    = (const float*)d_in[4];
    const float* W2    = (const float*)d_in[5];
    const float* b2    = (const float*)d_in[6];
    const float* Wf    = (const float*)d_in[7];
    const float* bfv   = (const float*)d_in[8];
    float* out = (float*)d_out;

    const int* src = eidx;
    const int* dst = eidx + N_EDGES;

    char* ws = (char*)d_ws;
    size_t off = 0;
    auto alloc = [&](size_t bytes) -> void* {
        void* p = ws + off;
        off = (off + bytes + 255) & ~(size_t)255;
        return p;
    };
    float*  bufA = (float*)alloc((size_t)N_NODES * DIM * sizeof(float));   // linear output
    float*  bufB = (float*)alloc((size_t)N_NODES * DIM * sizeof(float));   // aggregated
    __bf16* bufC = (__bf16*)alloc((size_t)N_NODES * DIM * sizeof(__bf16)); // bf16 activations
    float*  dis  = (float*)alloc((size_t)N_NODES * sizeof(float));
    __bf16* W2b  = (__bf16*)alloc((size_t)DIM * DIM * sizeof(__bf16));
    float*  sums = (float*)alloc((size_t)NGRAPH * DIM * sizeof(float));
    float*  cnts = (float*)alloc((size_t)NGRAPH * sizeof(float));

    const int B = 256;
    const int gN    = CEILDIV(N_NODES, B);
    const int gE    = CEILDIV(N_EDGES, B);
    const int gNF   = CEILDIV(N_NODES * DIM, B);
    const int gEW   = CEILDIV(N_EDGES, 8);            // 8 edges (waves) per block
    const int gW2   = CEILDIV(DIM * DIM, B);
    const int gWmma = CEILDIV(N_NODES / 16, 8);       // 8 strips (waves) per block
    const int gQ    = CEILDIV(NGRAPH * ACT, B);

    // normalization: deg -> dis (in-place)
    k_init_deg<<<gN, B, 0, stream>>>(dis);
    k_deg<<<gE, B, 0, stream>>>(dst, dis);
    k_dis<<<gN, B, 0, stream>>>(dis);

    // conv1
    k_lin1<<<gNF, B, 0, stream>>>(x, W1, bufA);
    k_selfinit<<<gNF, B, 0, stream>>>(bufA, dis, bufB);
    k_scatter<<<gEW, B, 0, stream>>>(src, dst, dis, bufA, bufB);
    k_relu_bf16<<<gNF, B, 0, stream>>>(bufB, b1, bufC);

    // conv2: WMMA GEMM then aggregation (no relu)
    k_packW2<<<gW2, B, 0, stream>>>(W2, W2b);
    k_gemm_wmma<<<gWmma, B, 0, stream>>>(bufC, W2b, bufA);
    k_selfinit<<<gNF, B, 0, stream>>>(bufA, dis, bufB);
    k_scatter<<<gEW, B, 0, stream>>>(src, dst, dis, bufA, bufB);

    // mean pool (+b2 folded in) and output head
    k_zero_pool<<<CEILDIV(NGRAPH * DIM, B), B, 0, stream>>>(sums, cnts);
    k_pool<<<gNF, B, 0, stream>>>(bufB, b2, batch, sums, cnts);
    k_final<<<gQ, B, 0, stream>>>(sums, cnts, Wf, bfv, out);
}
